// Attention_40561671144003
// MI455X (gfx1250) — matrix operand order repaired and checked by reading, not compiled, not run on hardware
//
#include <hip/hip_runtime.h>
#include <hip/hip_bf16.h>

// ---------- types ----------
typedef __bf16 bf16_t;
typedef __attribute__((ext_vector_type(8)))  bf16_t v8bf;
typedef __attribute__((ext_vector_type(16))) bf16_t v16bf;
typedef __attribute__((ext_vector_type(8)))  float  v8f;

#define WMMA_BF16(a, b, c) \
  __builtin_amdgcn_wmma_f32_16x16x32_bf16(false, (a), false, (b), (short)0, (c), false, false)

__device__ __forceinline__ bf16_t f2bf(float f) {
  unsigned u = __builtin_bit_cast(unsigned, f);
  unsigned r = (u + 0x7FFFu + ((u >> 16) & 1u)) >> 16;
  unsigned short s = (unsigned short)r;
  return __builtin_bit_cast(bf16_t, s);
}

// A-matrix 16x32 bf16 fragment (ISA 7.12.2): lane m = lane&15,
// klo = (lane<16)?0:8 ; VGPR0-3 hold K=klo..klo+7, VGPR4-7 hold K=16+klo..+7.
__device__ __forceinline__ v16bf load_a_frag(const bf16_t* base, int stride, int lane) {
  int m   = lane & 15;
  int klo = (lane < 16) ? 0 : 8;
  const bf16_t* p = base + m * stride + klo;
  v8bf lo = *(const v8bf*)p;          // 16B
  v8bf hi = *(const v8bf*)(p + 16);   // 16B
  return __builtin_shufflevector(lo, hi, 0,1,2,3,4,5,6,7,8,9,10,11,12,13,14,15);
}

// B-matrix 32x16 bf16 fragment: lane n = lane&15, klo = (lane<16)?0:8;
// elements 0-7 hold K=klo..klo+7, elements 8-15 hold K=16+klo..+7. `base` points at
// row n0 of B^T (operand stored K-major per output column).
__device__ __forceinline__ v16bf load_b_frag(const bf16_t* base, int stride, int lane) {
  int n    = lane & 15;
  int klo  = (lane < 16) ? 0 : 8;
  const bf16_t* p = base + n * stride + klo;
  v8bf lo = *(const v8bf*)p;          // 16B
  v8bf hi = *(const v8bf*)(p + 16);   // 16B
  return __builtin_shufflevector(lo, hi, 0,1,2,3,4,5,6,7,8,9,10,11,12,13,14,15);
}

// ---------- K0: precision conversion ----------
__global__ __launch_bounds__(256) void cvt_bf16(const float* __restrict__ s,
                                                bf16_t* __restrict__ d, int n) {
  int i = blockIdx.x * 256 + threadIdx.x;
  if (i < n) d[i] = f2bf(s[i]);
}

// x = concat(query, value) along channels, bf16. [4096,1024]
__global__ __launch_bounds__(256) void build_x(const float* __restrict__ q,
                                               const float* __restrict__ v,
                                               bf16_t* __restrict__ x) {
  int i = blockIdx.x * 256 + threadIdx.x;
  if (i < 4096 * 1024) {
    int n = i >> 10, c = i & 1023;
    float val = (c < 512) ? q[n * 512 + c] : v[n * 512 + (c - 512)];
    x[i] = f2bf(val);
  }
}

// scale folded into Q at store time: d^-0.5 * log2(e), softmax done base-2
#define QSCALE (0.125f * 1.44269504088896340736f)

// GEMM wave-tile stages: 32x64 per wave (2 A frags, 4 B frags, 8 C frags).
// Ping-pong double buffering with the K loop unrolled by 2 -> zero register
// rotation moves, MMAs wait only on loads issued a half-iteration earlier.
#define LOAD_STAGE(aa, bb, kk)                                            \
  do {                                                                    \
    aa[0] = load_a_frag(A0 + (kk), K, lane);                              \
    aa[1] = load_a_frag(A1 + (kk), K, lane);                              \
    _Pragma("unroll") for (int f = 0; f < 4; ++f)                         \
        bb[f] = load_b_frag(Wc + f * 16 * K + (kk), K, lane);             \
  } while (0)

#define MMA_STAGE(aa, bb)                                                 \
  do {                                                                    \
    _Pragma("unroll") for (int f = 0; f < 4; ++f) {                       \
      acc[0][f] = WMMA_BF16(aa[0], bb[f], acc[0][f]);                     \
      acc[1][f] = WMMA_BF16(aa[1], bb[f], acc[1][f]);                     \
    }                                                                     \
  } while (0)

// ---------- K1: QKV GEMM  qkv = x @ w_qkv^T ----------
__global__ __launch_bounds__(256) void qkv_gemm(const bf16_t* __restrict__ X,
                                                const bf16_t* __restrict__ W,
                                                bf16_t* __restrict__ qws,
                                                bf16_t* __restrict__ kws,
                                                bf16_t* __restrict__ vws,
                                                bf16_t* __restrict__ vt) {
  const int K = 1024;
  int lane = threadIdx.x & 31;
  int wave = threadIdx.x >> 5;
  int r0 = blockIdx.x * 256 + wave * 32;  // token rows (two 16-row sub-tiles)
  int c0 = blockIdx.y * 64;               // qkv channels

  const bf16_t* A0 = X + r0 * K;
  const bf16_t* A1 = X + (r0 + 16) * K;
  const bf16_t* Wc = W + c0 * K;

  v8f acc[2][4] = {};
  v16bf a0[2], b0[4], a1[2], b1[4];
  LOAD_STAGE(a0, b0, 0);
  for (int k0 = 0; k0 < K; k0 += 64) {
    LOAD_STAGE(a1, b1, k0 + 32);
    MMA_STAGE(a0, b0);
    LOAD_STAGE(a0, b0, (k0 + 64) & (K - 1));  // final wrap: dead in-range load
    MMA_STAGE(a1, b1);
  }

  int mofs = (lane < 16) ? 0 : 8;
  int nl = lane & 15;
#pragma unroll
  for (int t = 0; t < 2; ++t) {
#pragma unroll
    for (int f = 0; f < 4; ++f) {
      int col = c0 + f * 16 + nl;
      int cc = col & 511, h = cc >> 6, d = cc & 63;
#pragma unroll
      for (int r = 0; r < 8; ++r) {
        int tok = r0 + t * 16 + r + mofs;
        float v = acc[t][f][r];
        int idx = (h * 4096 + tok) * 64 + d;
        if (col < 512) {
          qws[idx] = f2bf(v * QSCALE);           // fold softmax scale into Q
        } else if (col < 1024) {
          kws[idx] = f2bf(v);
        } else {
          bf16_t bv = f2bf(v);
          vws[idx] = bv;
          vt[(h * 64 + d) * 4096 + tok] = bv;    // d-major for P@V B-operand
        }
      }
    }
  }
}

// v.reshape(B,N,-1) with no transpose == flat memory reinterpretation of [H][N][64]
__global__ __launch_bounds__(256) void copy_vflat(const bf16_t* __restrict__ v,
                                                  bf16_t* __restrict__ pin) {
  int i = blockIdx.x * 256 + threadIdx.x;
  if (i < 4096 * 512) pin[(i >> 9) * 1024 + 512 + (i & 511)] = v[i];
}

// ---------- K2: flash attention ----------
// WG = 4 waves x 16 query rows = 64 rows per head-block. 32 keys per iteration.
// V B-fragments are loaded at the top of the iteration so they are in flight
// behind the softmax VALU/trans work.
__global__ __launch_bounds__(128) void flash_attn(const bf16_t* __restrict__ qws,
                                                  const bf16_t* __restrict__ kws,
                                                  const bf16_t* __restrict__ vt,
                                                  bf16_t* __restrict__ pin) {
  __shared__ bf16_t lds[4][16][40];  // 16x32 P tile per wave, 80B row stride (16B aligned)
  int lane = threadIdx.x & 31;
  int wave = threadIdx.x >> 5;
  int h  = blockIdx.x >> 6;
  int n0 = (blockIdx.x & 63) * 64 + wave * 16;

  const bf16_t* qh  = qws + h * 4096 * 64;
  const bf16_t* kh  = kws + h * 4096 * 64;
  const bf16_t* vth = vt  + h * 64 * 4096;

  // Q A-fragments resident for the whole K-loop (K = d = 64 -> two 32-chunks)
  v16bf qa0 = load_a_frag(qh + n0 * 64 + 0,  64, lane);
  v16bf qa1 = load_a_frag(qh + n0 * 64 + 32, 64, lane);

  v8f o[4] = {};
  float mrow[8], lrow[8];
#pragma unroll
  for (int r = 0; r < 8; ++r) { mrow[r] = -3.0e38f; lrow[r] = 0.f; }

  for (int j0 = 0; j0 < 4096; j0 += 32) {
    // issue ALL loads for this iteration up front (K tiles + V tiles)
    v16bf kb0 = load_b_frag(kh + (j0     ) * 64 + 0,  64, lane);
    v16bf kb1 = load_b_frag(kh + (j0     ) * 64 + 32, 64, lane);
    v16bf kb2 = load_b_frag(kh + (j0 + 16) * 64 + 0,  64, lane);
    v16bf kb3 = load_b_frag(kh + (j0 + 16) * 64 + 32, 64, lane);
    v16bf vb[4];
#pragma unroll
    for (int f = 0; f < 4; ++f)
      vb[f] = load_b_frag(vth + (f * 16) * 4096 + j0, 4096, lane);

    // S = Q @ K^T (pre-scaled, base-2 logits)
    v8f s0 = {}, s1 = {};
    s0 = WMMA_BF16(qa0, kb0, s0);
    s0 = WMMA_BF16(qa1, kb1, s0);
    s1 = WMMA_BF16(qa0, kb2, s1);
    s1 = WMMA_BF16(qa1, kb3, s1);

    // Online softmax; C-frag rows live across 16-lane halves -> xor reduce {1,2,4,8}
#pragma unroll
    for (int r = 0; r < 8; ++r) {
      float mx = fmaxf(s0[r], s1[r]);
      mx = fmaxf(mx, __shfl_xor(mx, 1));
      mx = fmaxf(mx, __shfl_xor(mx, 2));
      mx = fmaxf(mx, __shfl_xor(mx, 4));
      mx = fmaxf(mx, __shfl_xor(mx, 8));
      float mnew = fmaxf(mrow[r], mx);
      float corr = exp2f(mrow[r] - mnew);      // native v_exp_f32
      float p0 = exp2f(s0[r] - mnew);
      float p1 = exp2f(s1[r] - mnew);
      float ps = p0 + p1;
      ps += __shfl_xor(ps, 1);
      ps += __shfl_xor(ps, 2);
      ps += __shfl_xor(ps, 4);
      ps += __shfl_xor(ps, 8);
      lrow[r] = lrow[r] * corr + ps;
      mrow[r] = mnew;
      int m = r + ((lane >= 16) ? 8 : 0);
      int c = lane & 15;
      lds[wave][m][c]      = f2bf(p0);   // C-frag -> row-major P tile in LDS
      lds[wave][m][c + 16] = f2bf(p1);
      o[0][r] *= corr; o[1][r] *= corr; o[2][r] *= corr; o[3][r] *= corr;
    }
    // per-wave LDS RAW: DS ops are in-order per wave; wait for store data visibility
    asm volatile("s_wait_dscnt 0" ::: "memory");
    v16bf pa = load_a_frag(&lds[wave][0][0], 40, lane);  // 16x32 A-frag of P
#pragma unroll
    for (int f = 0; f < 4; ++f) o[f] = WMMA_BF16(pa, vb[f], o[f]);
  }

  // epilogue: O/l -> pin[:, h*64 + d] (bf16 input of proj GEMM)
  int nl = lane & 15;
#pragma unroll
  for (int r = 0; r < 8; ++r) {
    float inv = 1.0f / lrow[r];
    int tok = n0 + r + ((lane >= 16) ? 8 : 0);
#pragma unroll
    for (int f = 0; f < 4; ++f)
      pin[tok * 1024 + h * 64 + f * 16 + nl] = f2bf(o[f][r] * inv);
  }
}

// ---------- K3: projection GEMM + bias + output split ----------
__global__ __launch_bounds__(256) void proj_gemm(const bf16_t* __restrict__ P,
                                                 const bf16_t* __restrict__ W,
                                                 const float* __restrict__ bias,
                                                 float* __restrict__ out) {
  const int K = 1024;
  int lane = threadIdx.x & 31;
  int wave = threadIdx.x >> 5;
  int r0 = blockIdx.x * 256 + wave * 32;
  int c0 = blockIdx.y * 64;

  const bf16_t* A0 = P + r0 * K;
  const bf16_t* A1 = P + (r0 + 16) * K;
  const bf16_t* Wc = W + c0 * K;

  v8f acc[2][4] = {};
  v16bf a0[2], b0[4], a1[2], b1[4];
  LOAD_STAGE(a0, b0, 0);
  for (int k0 = 0; k0 < K; k0 += 64) {
    LOAD_STAGE(a1, b1, k0 + 32);
    MMA_STAGE(a0, b0);
    LOAD_STAGE(a0, b0, (k0 + 64) & (K - 1));
    MMA_STAGE(a1, b1);
  }

  int mofs = (lane < 16) ? 0 : 8;
  int nl = lane & 15;
#pragma unroll
  for (int t = 0; t < 2; ++t) {
#pragma unroll
    for (int f = 0; f < 4; ++f) {
      int col = c0 + f * 16 + nl;
      float bb = bias[col];
#pragma unroll
      for (int r = 0; r < 8; ++r) {
        int tok = r0 + t * 16 + r + mofs;
        float v = acc[t][f][r] + bb;
        if (col < 512) out[tok * 512 + col] = v;                       // y[0]
        else           out[4096 * 512 + tok * 512 + (col - 512)] = v;  // y[1]
      }
    }
  }
}

// ---------- host ----------
extern "C" void kernel_launch(void* const* d_in, const int* in_sizes, int n_in,
                              void* d_out, int out_size, void* d_ws, size_t ws_size,
                              hipStream_t stream) {
  const float* q     = (const float*)d_in[0];
  const float* v     = (const float*)d_in[1];
  const float* wqkv  = (const float*)d_in[2];
  const float* wproj = (const float*)d_in[3];
  const float* bproj = (const float*)d_in[4];
  float* out = (float*)d_out;

  bf16_t* xb     = (bf16_t*)d_ws;               // 4096*1024
  bf16_t* wqkvb  = xb     + 4096 * 1024;        // 1536*1024
  bf16_t* wprojb = wqkvb  + 1536 * 1024;        // 1024*1024
  bf16_t* qws    = wprojb + 1024 * 1024;        // 8*4096*64 each:
  bf16_t* kws    = qws    + 8 * 4096 * 64;
  bf16_t* vws    = kws    + 8 * 4096 * 64;
  bf16_t* vt     = vws    + 8 * 4096 * 64;
  bf16_t* pin    = vt     + 8 * 4096 * 64;      // 4096*1024  (~37 MB total)

  build_x<<<(4096 * 1024 + 255) / 256, 256, 0, stream>>>(q, v, xb);
  cvt_bf16<<<(1536 * 1024 + 255) / 256, 256, 0, stream>>>(wqkv, wqkvb, 1536 * 1024);
  cvt_bf16<<<(1024 * 1024 + 255) / 256, 256, 0, stream>>>(wproj, wprojb, 1024 * 1024);

  qkv_gemm<<<dim3(16, 24), 256, 0, stream>>>(xb, wqkvb, qws, kws, vws, vt);
  copy_vflat<<<(4096 * 512 + 255) / 256, 256, 0, stream>>>(vws, pin);
  flash_attn<<<512, 128, 0, stream>>>(qws, kws, vt, pin);
  proj_gemm<<<dim3(16, 16), 256, 0, stream>>>(pin, wprojb, bproj, out);
}
